// SpectralMixer_43628277793670
// MI455X (gfx1250) — matrix operand order repaired and checked by reading, not compiled, not run on hardware
//
#include <hip/hip_runtime.h>
#include <hip/hip_bf16.h>

// ---------------------------------------------------------------------------
// SpectralMixer for MI455X (gfx1250, wave32, WMMA).
//   B=4, L=4096, d=1024, kconv=4, kmlp=128, pdim=64, FFT=8192, GROUP=128
// ---------------------------------------------------------------------------

typedef _Float16 v16h __attribute__((ext_vector_type(16)));
typedef _Float16 v8h  __attribute__((ext_vector_type(8)));
typedef float    v8f  __attribute__((ext_vector_type(8)));

#define SM_B     4
#define SM_L     4096
#define SM_D     1024
#define SM_K     1024
#define SM_M     (SM_B * SM_L)
#define SM_FFT   8192
#define SM_LOG2F 13

// ---------------------------------------------------------------------------
// 1) Fake-quantize a (d x d) fp32 weight to dequantized fp16 (group=128).
//    One 128-thread block per group of 128 consecutive elements.
// ---------------------------------------------------------------------------
__global__ void sm_quant(const float* __restrict__ w, _Float16* __restrict__ out) {
    __shared__ float red[128];
    const int g = blockIdx.x;
    const int i = threadIdx.x;
    const float v = w[(size_t)g * 128 + i];
    red[i] = fabsf(v);
    __syncthreads();
    for (int s = 64; s > 0; s >>= 1) {
        if (i < s) red[i] = fmaxf(red[i], red[i + s]);
        __syncthreads();
    }
    const float scale = fmaxf(red[0], 1e-10f) / 7.0f;
    float q = rintf(v / scale);               // round-half-even like jnp.round
    q = fminf(fmaxf(q, -8.0f), 7.0f);
    out[(size_t)g * 128 + i] = (_Float16)(q * scale);
}

// ---------------------------------------------------------------------------
// 2) Depthwise causal short conv (k=4), fp32 in -> fp16 out (GEMM A matrix).
//    y[b,t,c] = sum_{j=0..3} x[b, t-3+j, c] * w[c,0,j]
// ---------------------------------------------------------------------------
__global__ void sm_shortconv(const float* __restrict__ x,
                             const float* __restrict__ wconv,
                             _Float16* __restrict__ xl, int total) {
    int idx = blockIdx.x * blockDim.x + threadIdx.x;
    if (idx >= total) return;
    const int c = idx & (SM_D - 1);
    const int t = (idx >> 10) & (SM_L - 1);
    const int b = idx >> 22;
    const float* wc = wconv + c * 4;
    float acc = 0.0f;
#pragma unroll
    for (int j = 0; j < 4; ++j) {
        const int ts = t - 3 + j;
        if (ts >= 0) acc += x[((size_t)b * SM_L + ts) * SM_D + c] * wc[j];
    }
    xl[idx] = (_Float16)acc;
}

// ---------------------------------------------------------------------------
// 3) Implicit kernel generator: pos-enc -> MLP -> decay. One block per t.
//    Writes h_t[c, t]  (d x L, fp32).
// ---------------------------------------------------------------------------
__global__ void sm_hmlp(const float* __restrict__ w1,   // (128,64)
                        const float* __restrict__ b1,   // (128,)
                        const float* __restrict__ w2,   // (1024,128)
                        const float* __restrict__ b2,   // (1024,)
                        const float* __restrict__ alpha_raw,
                        float* __restrict__ h_t) {
    __shared__ float pe[64];
    __shared__ float hid[128];
    const int t = blockIdx.x;
    const int tid = threadIdx.x;              // 128 threads
    if (tid < 32) {
        const float lt = logf(10000.0f) / 31.0f;
        const float invf = expf(-(float)tid * lt);
        const float e = (float)t * invf;
        pe[tid]      = sinf(e);
        pe[tid + 32] = cosf(e);
    }
    __syncthreads();
    float acc = b1[tid];
    const float* w1r = w1 + tid * 64;
#pragma unroll 8
    for (int i = 0; i < 64; ++i) acc += pe[i] * w1r[i];
    hid[tid] = fmaxf(acc, 0.0f);
    __syncthreads();
    for (int c = tid; c < SM_D; c += 128) {
        float a = b2[c];
        const float* wr = w2 + (size_t)c * 128;
#pragma unroll 8
        for (int j = 0; j < 128; ++j) a += hid[j] * wr[j];
        const float alpha = log1pf(expf(alpha_raw[c]));   // softplus
        const float dec = expf(-alpha * (float)t);
        h_t[(size_t)c * SM_L + t] = a * dec;
    }
}

// ---------------------------------------------------------------------------
// Shared radix-2 DIT stages (input in bit-reversed order -> natural order).
// N = 8192, 256 threads, dir = -1 forward, +1 inverse.
// ---------------------------------------------------------------------------
__device__ __forceinline__ void sm_fft_stages(float2* s, int tid, float dir) {
    for (int stage = 1; stage <= SM_LOG2F; ++stage) {
        const int m = 1 << stage;
        const int half = m >> 1;
        for (int idx = tid; idx < (SM_FFT >> 1); idx += 256) {
            const int grp = idx >> (stage - 1);
            const int pos = idx & (half - 1);
            const int base = grp * m + pos;
            const float ang = dir * 6.283185307179586f * (float)pos / (float)m;
            float sw, cw;
            sincosf(ang, &sw, &cw);
            const float2 u = s[base];
            const float2 v = s[base + half];
            const float2 tw = {v.x * cw - v.y * sw, v.x * sw + v.y * cw};
            s[base]        = {u.x + tw.x, u.y + tw.y};
            s[base + half] = {u.x - tw.x, u.y - tw.y};
        }
        __syncthreads();
    }
}

// ---------------------------------------------------------------------------
// 4) Forward FFT of h rows: H_f[c] = FFT_8192(pad(h_t[c])), full spectrum.
// ---------------------------------------------------------------------------
__global__ void sm_fft_h(const float* __restrict__ h_t, float2* __restrict__ H_f) {
    extern __shared__ float2 s[];
    const int c = blockIdx.x;
    const float* row = h_t + (size_t)c * SM_L;
    for (int i = threadIdx.x; i < SM_FFT; i += 256) {
        const int j = __brev((unsigned)i) >> (32 - SM_LOG2F);
        const float v = (j < SM_L) ? row[j] : 0.0f;
        s[i] = {v, 0.0f};
    }
    __syncthreads();
    sm_fft_stages(s, threadIdx.x, -1.0f);
    float2* dst = H_f + (size_t)c * SM_FFT;
    for (int i = threadIdx.x; i < SM_FFT; i += 256) dst[i] = s[i];
}

// ---------------------------------------------------------------------------
// 5) Fused FFT conv: y_t[row] = irfft(FFT(pad(v_t[row])) * H_f[c])[:L]
//    One workgroup per (b,c) row.
// ---------------------------------------------------------------------------
__global__ void sm_fft_conv(const float* __restrict__ v_t,
                            const float2* __restrict__ H_f,
                            float* __restrict__ y_t) {
    extern __shared__ float2 s[];
    const int row = blockIdx.x;               // b*d + c
    const int c = row & (SM_D - 1);
    const float* src = v_t + (size_t)row * SM_L;
    for (int i = threadIdx.x; i < SM_FFT; i += 256) {
        const int j = __brev((unsigned)i) >> (32 - SM_LOG2F);
        const float v = (j < SM_L) ? src[j] : 0.0f;
        s[i] = {v, 0.0f};
    }
    __syncthreads();
    sm_fft_stages(s, threadIdx.x, -1.0f);     // forward FFT (natural order out)
    const float2* H = H_f + (size_t)c * SM_FFT;
    for (int i = threadIdx.x; i < SM_FFT; i += 256) {
        const float2 a = s[i];
        const float2 h = H[i];
        s[i] = {a.x * h.x - a.y * h.y, a.x * h.y + a.y * h.x};
    }
    __syncthreads();
    // bit-reverse permute in place (disjoint swap pairs), then inverse stages
    for (int i = threadIdx.x; i < SM_FFT; i += 256) {
        const int j = __brev((unsigned)i) >> (32 - SM_LOG2F);
        if (j > i) { const float2 t = s[i]; s[i] = s[j]; s[j] = t; }
    }
    __syncthreads();
    sm_fft_stages(s, threadIdx.x, 1.0f);      // inverse FFT
    const float inv = 1.0f / (float)SM_FFT;
    float* dst = y_t + (size_t)row * SM_L;
    for (int i = threadIdx.x; i < SM_L; i += 256) dst[i] = s[i].x * inv;
}

// ---------------------------------------------------------------------------
// 6) WMMA GEMM: C = act( A(M x K, f16 row-major) @ W(N x K, f16 row-major)^T )
//    One wave -> one 16x16 C tile; 8 waves/block share an A row-tile.
//    mode 0: C[m*N+n] = v          (final output, fp32)
//    mode 1: C[m*N+n] = sigmoid(v) (gate, fp32)
//    mode 2: C[((b*N)+n)*L + t]    (transposed store for FFT rows, fp32)
// ---------------------------------------------------------------------------
__global__ void sm_gemm(const _Float16* __restrict__ A,
                        const _Float16* __restrict__ W,
                        float* __restrict__ C, int mode) {
    const int wave = threadIdx.x >> 5;
    const int lane = threadIdx.x & 31;
    const int lh = lane & 15;
    const int sel = lane >> 4;

    const int n0 = blockIdx.x * 128 + wave * 16;
    const int m0 = blockIdx.y * 16;

    const _Float16* Arow = A + (size_t)(m0 + lh) * SM_K;
    const _Float16* Wrow = W + (size_t)(n0 + lh) * SM_K;

    v8f acc = {};
    for (int k0 = 0; k0 < SM_K; k0 += 32) {
        // A fragment: lanes 0-15 row m0+lh K[k0..], lanes 16-31 same row,
        // per ISA 16-bit A layout (v0-3: K+8*sel, v4-7: K+16+8*sel)
        const v8h alo = *(const v8h*)(Arow + k0 + 8 * sel);
        const v8h ahi = *(const v8h*)(Arow + k0 + 16 + 8 * sel);
        const v16h a = __builtin_shufflevector(alo, ahi,
            0, 1, 2, 3, 4, 5, 6, 7, 8, 9, 10, 11, 12, 13, 14, 15);
        // B fragment: lane holds column n0+lh, same K layout as A
        const v8h blo = *(const v8h*)(Wrow + k0 + 8 * sel);
        const v8h bhi = *(const v8h*)(Wrow + k0 + 16 + 8 * sel);
        const v16h b = __builtin_shufflevector(blo, bhi,
            0, 1, 2, 3, 4, 5, 6, 7, 8, 9, 10, 11, 12, 13, 14, 15);
        if (k0 + 32 < SM_K) {
            __builtin_prefetch(Arow + k0 + 32, 0, 1);
            __builtin_prefetch(Wrow + k0 + 32, 0, 1);
        }
        acc = __builtin_amdgcn_wmma_f32_16x16x32_f16(
            false, a, false, b, (short)0, acc, false, false);
    }

    const int n = n0 + lh;
#pragma unroll
    for (int i = 0; i < 8; ++i) {
        const int m = m0 + i + 8 * sel;
        float v = acc[i];
        if (mode == 1) v = 1.0f / (1.0f + expf(-v));
        if (mode == 2) {
            const int b = m >> 12;            // m / L
            const int t = m & (SM_L - 1);
            C[((size_t)b * SM_D + n) * SM_L + t] = v;
        } else {
            C[(size_t)m * SM_D + n] = v;
        }
    }
}

// ---------------------------------------------------------------------------
// 7) a2 = f16( g[b,t,c] * y_t[b,c,t] )   (A matrix for the final GEMM)
// ---------------------------------------------------------------------------
__global__ void sm_mulcvt(const float* __restrict__ g,
                          const float* __restrict__ y_t,
                          _Float16* __restrict__ a2, int total) {
    int idx = blockIdx.x * blockDim.x + threadIdx.x;
    if (idx >= total) return;
    const int c = idx & (SM_D - 1);
    const int t = (idx >> 10) & (SM_L - 1);
    const int b = idx >> 22;
    const float yv = y_t[((size_t)b * SM_D + c) * SM_L + t];
    a2[idx] = (_Float16)(g[idx] * yv);
}

// ---------------------------------------------------------------------------
// Host-side orchestration (graph-capture safe: launches only).
// ---------------------------------------------------------------------------
extern "C" void kernel_launch(void* const* d_in, const int* in_sizes, int n_in,
                              void* d_out, int out_size, void* d_ws, size_t ws_size,
                              hipStream_t stream) {
    const float* x          = (const float*)d_in[0];
    const float* short_conv = (const float*)d_in[1];
    const float* w_v        = (const float*)d_in[2];
    const float* w_g        = (const float*)d_in[3];
    const float* w_out      = (const float*)d_in[4];
    const float* mlp_w1     = (const float*)d_in[5];
    const float* mlp_b1     = (const float*)d_in[6];
    const float* mlp_w2     = (const float*)d_in[7];
    const float* mlp_b2     = (const float*)d_in[8];
    const float* alpha_raw  = (const float*)d_in[9];

    char* ws = (char*)d_ws;
    size_t off = 0;
    auto carve = [&](size_t bytes) -> void* {
        void* p = ws + off;
        off += (bytes + 255) & ~(size_t)255;
        return p;
    };
    _Float16* wq_v   = (_Float16*)carve((size_t)SM_D * SM_K * sizeof(_Float16));
    _Float16* wq_g   = (_Float16*)carve((size_t)SM_D * SM_K * sizeof(_Float16));
    _Float16* wq_out = (_Float16*)carve((size_t)SM_D * SM_K * sizeof(_Float16));
    _Float16* x_loc  = (_Float16*)carve((size_t)SM_M * SM_D * sizeof(_Float16));
    float*    h_t    = (float*)   carve((size_t)SM_D * SM_L * sizeof(float));
    float2*   H_f    = (float2*)  carve((size_t)SM_D * SM_FFT * sizeof(float2));
    float*    v_t    = (float*)   carve((size_t)SM_B * SM_D * SM_L * sizeof(float));
    float*    gbuf   = (float*)   carve((size_t)SM_M * SM_D * sizeof(float));
    float*    y_t    = (float*)   carve((size_t)SM_B * SM_D * SM_L * sizeof(float));
    _Float16* a2     = x_loc;     // reuse: x_local dead after v/g GEMMs
    (void)in_sizes; (void)n_in; (void)out_size; (void)ws_size;

    const int total = SM_M * SM_D;            // 16,777,216
    const int ldsFFT = SM_FFT * sizeof(float2);
    const dim3 gemmGrid(SM_D / 128, SM_M / 16);

    // weights -> dequantized f16
    sm_quant<<<(SM_D * SM_K) / 128, 128, 0, stream>>>(w_v, wq_v);
    sm_quant<<<(SM_D * SM_K) / 128, 128, 0, stream>>>(w_g, wq_g);
    sm_quant<<<(SM_D * SM_K) / 128, 128, 0, stream>>>(w_out, wq_out);
    // short causal conv -> f16 A matrix
    sm_shortconv<<<total / 256, 256, 0, stream>>>(x, short_conv, x_loc, total);
    // implicit kernel + its spectrum
    sm_hmlp<<<SM_L, 128, 0, stream>>>(mlp_w1, mlp_b1, mlp_w2, mlp_b2, alpha_raw, h_t);
    sm_fft_h<<<SM_D, 256, ldsFFT, stream>>>(h_t, H_f);
    // v (stored transposed for FFT rows) and gate g
    sm_gemm<<<gemmGrid, 256, 0, stream>>>(x_loc, wq_v, v_t, 2);
    sm_gemm<<<gemmGrid, 256, 0, stream>>>(x_loc, wq_g, gbuf, 1);
    // FFT causal convolution
    sm_fft_conv<<<SM_B * SM_D, 256, ldsFFT, stream>>>(v_t, H_f, y_t);
    // gated combine -> f16, final projection -> d_out
    sm_mulcvt<<<total / 256, 256, 0, stream>>>(gbuf, y_t, a2, total);
    sm_gemm<<<gemmGrid, 256, 0, stream>>>(a2, wq_out, (float*)d_out, 0);
}
